// GQA_60000693125697
// MI455X (gfx1250) — matrix operand order repaired and checked by reading, not compiled, not run on hardware
//
#include <hip/hip_runtime.h>

typedef __bf16 bf16_t;
typedef __attribute__((ext_vector_type(16))) __bf16 v16bf;
typedef __attribute__((ext_vector_type(8)))  __bf16 v8bf;
typedef __attribute__((ext_vector_type(8)))  float  v8f;
typedef __attribute__((ext_vector_type(4)))  int    v4i;

#define EMBED 2048
#define NHEAD 16
#define KVHEAD 2
#define HDIM 128
#define KVDIM 256
#define BSZ 2
#define TSZ 2048
#define MTOK (BSZ * TSZ)   // 4096 token rows

// GEMM block tiling
#define BM 128
#define BN 128
#define BK 32

#if defined(__has_builtin)
#  if __has_builtin(__builtin_amdgcn_global_load_async_to_lds_b128)
#    define HAVE_ASYNC_LDS 1
#  endif
#  if __has_builtin(__builtin_amdgcn_s_wait_asynccnt)
#    define HAVE_WAIT_ASYNC_BUILTIN 1
#  endif
#endif
#ifndef HAVE_ASYNC_LDS
#  define HAVE_ASYNC_LDS 0
#endif
#ifndef HAVE_WAIT_ASYNC_BUILTIN
#  define HAVE_WAIT_ASYNC_BUILTIN 0
#endif

typedef __attribute__((address_space(1))) v4i gv4i_t;   // global b128 chunk
typedef __attribute__((address_space(3))) v4i lv4i_t;   // LDS b128 chunk

union AFrag { v16bf v; v8bf h[2]; };

static __device__ __forceinline__ v8f wmma_bf16(v16bf a, v16bf b, v8f c) {
    // D = A(16x32 bf16) * B(32x16 bf16) + C(16x16 f32)
    return __builtin_amdgcn_wmma_f32_16x16x32_bf16(
        /*neg_a=*/false, a, /*neg_b=*/false, b,
        /*c_mod=*/(short)0, c, /*reuse_a=*/false, /*reuse_b=*/false);
}

static __device__ __forceinline__ void async_copy_b128(const bf16_t* g, bf16_t* l) {
#if HAVE_ASYNC_LDS
    __builtin_amdgcn_global_load_async_to_lds_b128(
        (gv4i_t*)(const void*)g,
        (lv4i_t*)(void*)l, 0, 0);
#else
    *reinterpret_cast<v8bf*>(l) = *reinterpret_cast<const v8bf*>(g);
#endif
}

static __device__ __forceinline__ void wait_async_done() {
#if HAVE_ASYNC_LDS
#  if HAVE_WAIT_ASYNC_BUILTIN
    __builtin_amdgcn_s_wait_asynccnt(0);
#  else
    asm volatile("s_wait_asynccnt 0" ::: "memory");
#  endif
#endif
}

// ---------------------------------------------------------------------------
// fp32 -> bf16 elementwise conversion
// ---------------------------------------------------------------------------
__global__ void __launch_bounds__(256)
cvt_f32_bf16_kernel(const float* __restrict__ in, bf16_t* __restrict__ out, int n) {
    int i = blockIdx.x * blockDim.x + threadIdx.x;
    if (i < n) out[i] = (bf16_t)in[i];
}

// v [B, T, KVDIM] -> vt [B, KVDIM, T]
__global__ void __launch_bounds__(256)
transpose_v_kernel(const bf16_t* __restrict__ v, bf16_t* __restrict__ vt) {
    int i = blockIdx.x * blockDim.x + threadIdx.x;
    int total = BSZ * TSZ * KVDIM;
    if (i >= total) return;
    int d = i % KVDIM;
    int t = (i / KVDIM) % TSZ;
    int b = i / (KVDIM * TSZ);
    vt[((size_t)b * KVDIM + d) * TSZ + t] = v[i];
}

// ---------------------------------------------------------------------------
// Blocked GEMM: out[M,N] = A[M,K](bf16) * W[N,K]^T(bf16) + bias[N]
// 256 threads = 8 waves. Block tile 128x128; wave tile 64x32 (4x2 WMMA tiles).
// A/W panels staged through double-buffered LDS with async-to-LDS copies.
// ---------------------------------------------------------------------------
template <bool OUT_F32>
__global__ void __launch_bounds__(256)
wmma_gemm_bias_kernel(const bf16_t* __restrict__ A,
                      const bf16_t* __restrict__ W,
                      const float*  __restrict__ bias,
                      void* __restrict__ outp,
                      int M, int N, int K) {
    __shared__ bf16_t As[2][BM * BK];   // [row][k] row-major, 8KB per buffer
    __shared__ bf16_t Bs[2][BN * BK];

    const int tid  = threadIdx.x;
    const int lane = tid & 31;
    const int wave = tid >> 5;
    const int lh   = lane & 15;
    const int half = lane >> 4;
    const int aoff = half << 4;   // A frag: lanes 0-15 K in {0..15}, 16-31 in {16..31}
    const int boff = half << 4;   // B frag: contiguous 16 K at {0,16}

    const int nblk = N / BN;
    const int mb   = (blockIdx.x / nblk) * BM;
    const int nb   = (blockIdx.x % nblk) * BN;

    // wave tile inside the block: 2 rows x 4 cols of waves
    const int wm = (wave >> 2) * 64;   // 0 or 64
    const int wn = (wave & 3) * 32;    // 0,32,64,96

    // stage one BK-chunk of the A and W panels (each: 128 rows x 64B = 512 b128)
    auto stage = [&](int buf, int k0) {
#pragma unroll
        for (int i = 0; i < 2; ++i) {
            int f   = tid + i * 256;        // 0..511
            int row = f >> 2;
            int c   = (f & 3) << 3;         // element offset within the 32-K chunk
            async_copy_b128(A + (size_t)(mb + row) * K + k0 + c, &As[buf][row * BK + c]);
            async_copy_b128(W + (size_t)(nb + row) * K + k0 + c, &Bs[buf][row * BK + c]);
        }
    };

    v8f acc[4][2];
#pragma unroll
    for (int mt = 0; mt < 4; ++mt)
#pragma unroll
        for (int nt = 0; nt < 2; ++nt) acc[mt][nt] = (v8f){};

    const int nchunks = K / BK;
    stage(0, 0);
    for (int kc = 0; kc < nchunks; ++kc) {
        const int buf = kc & 1;
        wait_async_done();
        __syncthreads();                       // staged panels visible to all waves
        if (kc + 1 < nchunks) stage(buf ^ 1, (kc + 1) * BK);

        // B fragments for this wave's two N tiles
        v16bf bf0 = *reinterpret_cast<const v16bf*>(&Bs[buf][(wn + lh) * BK + boff]);
        v16bf bf1 = *reinterpret_cast<const v16bf*>(&Bs[buf][(wn + 16 + lh) * BK + boff]);
#pragma unroll
        for (int mt = 0; mt < 4; ++mt) {
            AFrag a;
            const bf16_t* p = &As[buf][(wm + mt * 16 + lh) * BK];
            a.h[0] = *reinterpret_cast<const v8bf*>(p + aoff);
            a.h[1] = *reinterpret_cast<const v8bf*>(p + aoff + 8);
            acc[mt][0] = wmma_bf16(a.v, bf0, acc[mt][0]);
            acc[mt][1] = wmma_bf16(a.v, bf1, acc[mt][1]);
        }
        __syncthreads();                       // all reads done before next overwrite
    }

    // epilogue: bias + store (C layout: row = r + 8*half, col = lh)
#pragma unroll
    for (int nt = 0; nt < 2; ++nt) {
        const int col = nb + wn + nt * 16 + lh;
        const float bv = bias[col];
#pragma unroll
        for (int mt = 0; mt < 4; ++mt) {
            const int mbase = mb + wm + mt * 16 + (half << 3);
            if (OUT_F32) {
                float* out = (float*)outp;
#pragma unroll
                for (int r = 0; r < 8; ++r)
                    out[(size_t)(mbase + r) * N + col] = acc[mt][nt][r] + bv;
            } else {
                bf16_t* out = (bf16_t*)outp;
#pragma unroll
                for (int r = 0; r < 8; ++r)
                    out[(size_t)(mbase + r) * N + col] = (bf16_t)(acc[mt][nt][r] + bv);
            }
        }
    }
}

// ---------------------------------------------------------------------------
// Flash attention (non-causal, GQA). One wave = 16 query rows x full d=128.
// q [B*T, EMBED], k [B*T, KVDIM], vt [B, KVDIM, T], ctx [B*T, EMBED]
// ---------------------------------------------------------------------------
__global__ void __launch_bounds__(128)
flash_attn_kernel(const bf16_t* __restrict__ q,
                  const bf16_t* __restrict__ k,
                  const bf16_t* __restrict__ vt,
                  bf16_t* __restrict__ ctx) {
    __shared__ bf16_t lds_p[4][16 * 32];   // per-wave P staging tile

    const int lane = threadIdx.x & 31;
    const int wave = threadIdx.x >> 5;
    const int tile = blockIdx.x * 4 + wave;       // over B*NHEAD*(T/16)
    const int qtiles = TSZ >> 4;                  // 128
    if (tile >= BSZ * NHEAD * qtiles) return;

    const int m0  = (tile % qtiles) << 4;
    const int h   = (tile / qtiles) % NHEAD;
    const int b   = tile / (qtiles * NHEAD);
    const int kvh = h / (NHEAD / KVHEAD);
    const float scale = 0.08838834764831845f;     // 1/sqrt(128)

    const int lh   = lane & 15;
    const int half = lane >> 4;
    const int aoff = half << 4;
    const int boff = half << 4;

    // Q fragments: 4 chunks of K=32 covering d=128 (A-matrix layout)
    AFrag qf[4];
    {
        const bf16_t* qp = q + (size_t)(b * TSZ + m0 + lh) * EMBED + h * HDIM;
#pragma unroll
        for (int c = 0; c < 4; ++c) {
            qf[c].h[0] = *reinterpret_cast<const v8bf*>(qp + c * 32 + aoff);
            qf[c].h[1] = *reinterpret_cast<const v8bf*>(qp + c * 32 + aoff + 8);
        }
    }

    v8f oacc[8];
#pragma unroll
    for (int t = 0; t < 8; ++t) oacc[t] = (v8f){};
    float m_i[8], l_i[8];
#pragma unroll
    for (int r = 0; r < 8; ++r) { m_i[r] = -1e30f; l_i[r] = 0.0f; }

    bf16_t* myp = &lds_p[wave][0];

    for (int j0 = 0; j0 < TSZ; j0 += 32) {
        // ---- scores S = Q * K^T for keys j0..j0+31 (two 16-key tiles) ----
        v8f s0 = {}, s1 = {};
        {
            const bf16_t* kp0 =
                k + (size_t)(b * TSZ + j0 + lh) * KVDIM + kvh * HDIM + boff;
            const bf16_t* kp1 = kp0 + (size_t)16 * KVDIM;
#pragma unroll
            for (int c = 0; c < 4; ++c) {
                v16bf kb0 = *reinterpret_cast<const v16bf*>(kp0 + c * 32);
                s0 = wmma_bf16(qf[c].v, kb0, s0);
                v16bf kb1 = *reinterpret_cast<const v16bf*>(kp1 + c * 32);
                s1 = wmma_bf16(qf[c].v, kb1, s1);
            }
        }
        // ---- online softmax over these 32 keys ----
#pragma unroll
        for (int r = 0; r < 8; ++r) {
            float a0 = s0[r] * scale;
            float a1 = s1[r] * scale;
            float mx = fmaxf(a0, a1);
            mx = fmaxf(mx, __shfl_xor(mx, 1));
            mx = fmaxf(mx, __shfl_xor(mx, 2));
            mx = fmaxf(mx, __shfl_xor(mx, 4));
            mx = fmaxf(mx, __shfl_xor(mx, 8));
            float newm = fmaxf(m_i[r], mx);
            float corr = __expf(m_i[r] - newm);
            float p0 = __expf(a0 - newm);
            float p1 = __expf(a1 - newm);
            float rs = p0 + p1;
            rs += __shfl_xor(rs, 1);
            rs += __shfl_xor(rs, 2);
            rs += __shfl_xor(rs, 4);
            rs += __shfl_xor(rs, 8);
            l_i[r] = l_i[r] * corr + rs;
            m_i[r] = newm;
#pragma unroll
            for (int t = 0; t < 8; ++t) oacc[t][r] *= corr;
            // stash P tile row-major [16][32] in LDS (C layout -> memory)
            int row = r + (half << 3);
            myp[row * 32 + lh]      = (bf16_t)p0;
            myp[row * 32 + lh + 16] = (bf16_t)p1;
        }
        // same-wave LDS RAW: wait out the DS stores before the frag reload
        asm volatile("s_wait_dscnt 0" ::: "memory");
        // reload P as an A fragment (16x32)
        AFrag pf;
        pf.h[0] = *reinterpret_cast<const v8bf*>(myp + lh * 32 + aoff);
        pf.h[1] = *reinterpret_cast<const v8bf*>(myp + lh * 32 + aoff + 8);
        // ---- O += P * V  (8 WMMAs over d) ----
        const bf16_t* vp =
            vt + ((size_t)b * KVDIM + kvh * HDIM + lh) * TSZ + j0 + boff;
#pragma unroll
        for (int t = 0; t < 8; ++t) {
            v16bf vb = *reinterpret_cast<const v16bf*>(vp + (size_t)t * 16 * TSZ);
            oacc[t] = wmma_bf16(pf.v, vb, oacc[t]);
        }
    }

    // ---- finalize: divide by row sums and store context (bf16) ----
    const int mb = half << 3;
    bf16_t* cp = ctx + (size_t)(b * TSZ + m0 + mb) * EMBED + h * HDIM;
#pragma unroll
    for (int r = 0; r < 8; ++r) {
        float inv = 1.0f / l_i[r];
#pragma unroll
        for (int t = 0; t < 8; ++t)
            cp[(size_t)r * EMBED + t * 16 + lh] = (bf16_t)(oacc[t][r] * inv);
    }
}

// ---------------------------------------------------------------------------
extern "C" void kernel_launch(void* const* d_in, const int* in_sizes, int n_in,
                              void* d_out, int out_size, void* d_ws, size_t ws_size,
                              hipStream_t stream) {
    const float* x   = (const float*)d_in[0];
    const float* w_q = (const float*)d_in[1];
    const float* b_q = (const float*)d_in[2];
    const float* w_k = (const float*)d_in[3];
    const float* b_k = (const float*)d_in[4];
    const float* w_v = (const float*)d_in[5];
    const float* b_v = (const float*)d_in[6];
    const float* w_o = (const float*)d_in[7];
    const float* b_o = (const float*)d_in[8];
    float* out = (float*)d_out;

    // workspace carve-up (bf16 buffers), 256B aligned
    char* ws = (char*)d_ws;
    size_t off = 0;
    auto carve = [&](size_t elems) {
        bf16_t* p = (bf16_t*)(ws + off);
        off += (elems * sizeof(bf16_t) + 255) & ~(size_t)255;
        return p;
    };
    bf16_t* x_b   = carve((size_t)MTOK * EMBED);
    bf16_t* wq_b  = carve((size_t)EMBED * EMBED);
    bf16_t* wk_b  = carve((size_t)KVDIM * EMBED);
    bf16_t* wv_b  = carve((size_t)KVDIM * EMBED);
    bf16_t* wo_b  = carve((size_t)EMBED * EMBED);
    bf16_t* q_b   = carve((size_t)MTOK * EMBED);
    bf16_t* k_b   = carve((size_t)MTOK * KVDIM);
    bf16_t* v_b   = carve((size_t)MTOK * KVDIM);
    bf16_t* vt_b  = carve((size_t)BSZ * KVDIM * TSZ);
    bf16_t* ctx_b = carve((size_t)MTOK * EMBED);
    (void)ws_size; (void)in_sizes; (void)n_in; (void)out_size;

    auto cvt = [&](const float* src, bf16_t* dst, int n) {
        cvt_f32_bf16_kernel<<<(n + 255) / 256, 256, 0, stream>>>(src, dst, n);
    };
    cvt(x,   x_b,  MTOK * EMBED);
    cvt(w_q, wq_b, EMBED * EMBED);
    cvt(w_k, wk_b, KVDIM * EMBED);
    cvt(w_v, wv_b, KVDIM * EMBED);
    cvt(w_o, wo_b, EMBED * EMBED);

    // projections: 128x128 block tile per 256-thread workgroup
    {
        int blocks = (MTOK / BM) * (EMBED / BN);
        wmma_gemm_bias_kernel<false><<<blocks, 256, 0, stream>>>(
            x_b, wq_b, b_q, q_b, MTOK, EMBED, EMBED);
    }
    {
        int blocks = (MTOK / BM) * (KVDIM / BN);
        wmma_gemm_bias_kernel<false><<<blocks, 256, 0, stream>>>(
            x_b, wk_b, b_k, k_b, MTOK, KVDIM, EMBED);
        wmma_gemm_bias_kernel<false><<<blocks, 256, 0, stream>>>(
            x_b, wv_b, b_v, v_b, MTOK, KVDIM, EMBED);
    }
    {
        int n = BSZ * TSZ * KVDIM;
        transpose_v_kernel<<<(n + 255) / 256, 256, 0, stream>>>(v_b, vt_b);
    }
    {
        int tiles = BSZ * NHEAD * (TSZ / 16);   // 4096 waves
        flash_attn_kernel<<<tiles / 4, 128, 0, stream>>>(q_b, k_b, vt_b, ctx_b);
    }
    {
        int blocks = (MTOK / BM) * (EMBED / BN);
        wmma_gemm_bias_kernel<true><<<blocks, 256, 0, stream>>>(
            ctx_b, wo_b, b_o, out, MTOK, EMBED, EMBED);
    }
}
